// ModulatedConv2d_51573967290883
// MI455X (gfx1250) — matrix operand order repaired and checked by reading, not compiled, not run on hardware
//
#include <hip/hip_runtime.h>

// ---------------------------------------------------------------------------
// ModulatedConv2d (StyleGAN2) as implicit GEMM on CDNA5 WMMA bf16.
//   B=8, CIN=64, COUT=64, H=W=512, K=3
//   Per batch:  Y[cout, hw] = sum_{k=cin*9} Wm[cout,k] * Patch[k, hw]
//   M=64 (cout), K=576 (tap*64+cin), N=H*W (pixels)
// Roofline: ~155 GFLOP vs ~1.07 GB minimum HBM traffic (~46 us at 23.3 TB/s)
// -> matrix-core bound; bf16 WMMA with f32 accumulate is the right path.
// ---------------------------------------------------------------------------

typedef __bf16 bf16;
typedef bf16  v16bf __attribute__((ext_vector_type(16)));
typedef float v8f   __attribute__((ext_vector_type(8)));
typedef unsigned int v4u __attribute__((ext_vector_type(4)));

#define B_    8
#define CIN_  64
#define COUT_ 64
#define H_    512
#define W_    512
#define HW_   (H_ * W_)

#define TILE_H 4
#define TILE_W 32
#define PATCH_ROWS 6           // TILE_H + 2
#define PATCH_COLS 34          // TILE_W + 2
#define PSTRIDE    72          // halves per pixel in LDS (64 + 8 pad -> conflict-free, 16B aligned)
#define NKSTEP     18          // 576 / 32

__device__ __forceinline__ unsigned short f2bf(float f) {
  // round-to-nearest-even fp32 -> bf16
  unsigned u = __float_as_uint(f);
  u += 0x7FFFu + ((u >> 16) & 1u);
  return (unsigned short)(u >> 16);
}

// ---------------------------------------------------------------------------
// Prep: modulate + demodulate the shared 64x64x3x3 kernel per sample, emit
// bf16 directly in a contiguous-K 16x32 A-fragment lane layout (B uses the same):
//   lanes 0-15  : M=lane,    halves 0-15 = K 0..15
//   lanes 16-31 : M=lane-16, halves 0-15 = K 16..31
// Fragment storage: afrag[(((b*4+mt)*18 + ks)*32 + lane)*16 + e]
// ---------------------------------------------------------------------------
__global__ void modconv_prep(const float* __restrict__ wmod,
                             const float* __restrict__ weight,
                             unsigned short* __restrict__ afrag) {
  int tid = blockIdx.x * blockDim.x + threadIdx.x;
  if (tid >= B_ * COUT_) return;
  int b = tid >> 6;
  int o = tid & 63;
  const float scale = 0.041666666666666664f;   // 1/sqrt(64*9)

  float ss = 1e-8f;
  for (int t = 0; t < 9; ++t)
    for (int c = 0; c < CIN_; ++c) {
      float v = weight[(o * CIN_ + c) * 9 + t] * wmod[b * CIN_ + c] * scale;
      ss += v * v;
    }
  float d = rsqrtf(ss);

  int mt = o >> 4;
  int M  = o & 15;
  for (int t = 0; t < 9; ++t)
    for (int c = 0; c < CIN_; ++c) {
      float v = weight[(o * CIN_ + c) * 9 + t] * wmod[b * CIN_ + c] * scale * d;
      int k  = t * CIN_ + c;       // K-index: tap-major, cin-minor
      int ks = k >> 5;
      int kk = k & 31;
      int lane, e;
      if (kk < 16) { lane = M;      e = kk;      }
      else         { lane = M + 16; e = kk - 16; }
      afrag[(((b * 4 + mt) * NKSTEP + ks) * 32 + lane) * 16 + e] = f2bf(v);
    }
}

// ---------------------------------------------------------------------------
// Implicit-GEMM conv. One block = 64 couts x (4h x 32w) pixels.
// 8 waves: wave w -> mt = w>>1 (16-cout tile), nset = (w&1)*4 (four 16-pixel
// N-tiles). Fully unrolled 18 K-steps: all LDS/global offsets become
// immediates; 4 B-fragment loads are grouped before their 4 WMMAs so DS
// latency overlaps matrix-op execution.
// ---------------------------------------------------------------------------
__global__ void __launch_bounds__(256)
modconv_gemm(const float* __restrict__ x,
             const unsigned short* __restrict__ afrag,
             float* __restrict__ out) {
  __shared__ unsigned short sx[PATCH_ROWS * PATCH_COLS * PSTRIDE];

  const int b  = blockIdx.z;
  const int h0 = blockIdx.y * TILE_H;
  const int w0 = blockIdx.x * TILE_W;
  const int tid = threadIdx.x;

  // ---- stage input patch (fp32 global -> bf16 LDS), zero-padded halo ----
  {
    const int p = tid;                       // 204 pixels, 256 threads
    int row = p / PATCH_COLS;
    int col = p - row * PATCH_COLS;
    int hh = h0 - 1 + row;
    int ww = w0 - 1 + col;
    bool valid = (p < PATCH_ROWS * PATCH_COLS) &&
                 ((unsigned)hh < (unsigned)H_) && ((unsigned)ww < (unsigned)W_);
    const float* gp = x + (size_t)b * CIN_ * HW_ + hh * W_ + ww;
    unsigned short* sp = &sx[p * PSTRIDE];
    if (p < PATCH_ROWS * PATCH_COLS) {
#pragma unroll 4
      for (int c = 0; c < CIN_; ++c) {
        float v = valid ? gp[(size_t)c * HW_] : 0.f;
        sp[c] = f2bf(v);
      }
    }
  }
  __syncthreads();

  const int wave = tid >> 5;
  const int lane = tid & 31;
  const int mt   = wave >> 1;        // cout tile 0..3
  const int nset = (wave & 1) * 4;   // first of four N-tiles
  const int col  = lane & 15;        // N within tile / C-D column
  const int hi   = lane >> 4;        // half-wave select

  v8f acc[4] = {};

  // Per-N-tile LDS base (lane-dependent); per-K-step delta is an immediate.
  const unsigned short* sb[4];
#pragma unroll
  for (int i = 0; i < 4; ++i) {
    const int nt = nset + i;
    sb[i] = &sx[(((nt >> 1) * PATCH_COLS) + ((nt & 1) << 4) + col) * PSTRIDE +
                (hi << 4)];
  }

  // A fragments: 32B per lane per K-step, contiguous, L2-hot across blocks.
  const unsigned short* ap =
      afrag + (((size_t)(b * 4 + mt) * NKSTEP) << 9) + (lane << 4);

#pragma unroll
  for (int ks = 0; ks < NKSTEP; ++ks) {
    const int tap = ks >> 1;
    const int kh  = tap / 3;
    const int kw  = tap - kh * 3;
    // wave-uniform, compile-time LDS delta for this K-step
    const int soff = (kh * PATCH_COLS + kw) * PSTRIDE + ((ks & 1) << 5);

    union { v4u q[2]; v16bf v; } A;
    {
      const v4u* aq = (const v4u*)(ap + (ks << 9));
      A.q[0] = aq[0];
      A.q[1] = aq[1];
    }

    union { v4u q[2]; v16bf v; } Bf[4];
#pragma unroll
    for (int i = 0; i < 4; ++i) {
      const v4u* bq = (const v4u*)(sb[i] + soff);
      Bf[i].q[0] = bq[0];
      Bf[i].q[1] = bq[1];
    }
#pragma unroll
    for (int i = 0; i < 4; ++i)
      acc[i] = __builtin_amdgcn_wmma_f32_16x16x32_bf16(
          false, A.v, false, Bf[i].v, (short)0, acc[i], false, false);
  }

  // ---- store: C/D layout -> lane L, vgpr j: M = j + 8*hi, N = L&15 ----
  const int cout0 = mt * 16 + hi * 8;
#pragma unroll
  for (int i = 0; i < 4; ++i) {
    const int nt = nset + i;
    const int oh = h0 + (nt >> 1);
    const int ow = w0 + ((nt & 1) << 4) + col;
    float* op = out + ((size_t)(b * COUT_ + cout0) * H_ + oh) * W_ + ow;
#pragma unroll
    for (int j = 0; j < 8; ++j)
      op[(size_t)j * HW_] = acc[i][j];
  }
}

// ---------------------------------------------------------------------------
extern "C" void kernel_launch(void* const* d_in, const int* in_sizes, int n_in,
                              void* d_out, int out_size, void* d_ws, size_t ws_size,
                              hipStream_t stream) {
  const float* x      = (const float*)d_in[0];   // [8,64,512,512]
  const float* wmod   = (const float*)d_in[1];   // [8,64]
  const float* weight = (const float*)d_in[2];   // [64,64,3,3]
  float* out = (float*)d_out;                    // [8,64,512,512]

  // A-fragment workspace: 8 * 4 * 18 * 512 halves = 589,824 bytes
  unsigned short* afrag = (unsigned short*)d_ws;

  modconv_prep<<<2, 256, 0, stream>>>(wmod, weight, afrag);
  modconv_gemm<<<dim3(W_ / TILE_W, H_ / TILE_H, B_), 256, 0, stream>>>(x, afrag, out);
}
